// SimCSE_47828755808773
// MI455X (gfx1250) — matrix operand order repaired and checked by reading, not compiled, not run on hardware
//
#include <hip/hip_runtime.h>
#include <hip/hip_bf16.h>

typedef __attribute__((ext_vector_type(16))) _Float16 v16h;
typedef __attribute__((ext_vector_type(8)))  _Float16 v8h;
typedef __attribute__((ext_vector_type(4)))  _Float16 v4h;
typedef __attribute__((ext_vector_type(8)))  float    v8f;
typedef __attribute__((ext_vector_type(4)))  unsigned int v4u;
typedef __attribute__((ext_vector_type(8)))  int      v8i;
typedef __attribute__((ext_vector_type(4)))  int      v4i;

#define BDIM   8192
#define DDIM   1024
#define INV_T  10.0f
#define EPSN   1e-8f
#define TILE_M 128
#define TILE_N 128
#define KSTEP  32
#define NKSTEP (DDIM / KSTEP)

// ---------------------------------------------------------------------------
// Phase 1: row-normalize X (scaled by 1/T) and Y into f16; zero accumulators.
// ---------------------------------------------------------------------------
__global__ __launch_bounds__(256) void normalize_rows(
    const float* __restrict__ X, const float* __restrict__ Y,
    _Float16* __restrict__ Xn, _Float16* __restrict__ Yn,
    float* __restrict__ rowsum, float* __restrict__ diag)
{
  const int row  = blockIdx.x;
  const int tid  = threadIdx.x;
  const int w    = tid >> 5;
  const int lane = tid & 31;

  const float4 xv = ((const float4*)(X + (size_t)row * DDIM))[tid];
  const float4 yv = ((const float4*)(Y + (size_t)row * DDIM))[tid];

  float ssx = xv.x*xv.x + xv.y*xv.y + xv.z*xv.z + xv.w*xv.w;
  float ssy = yv.x*yv.x + yv.y*yv.y + yv.z*yv.z + yv.w*yv.w;
#pragma unroll
  for (int m = 16; m >= 1; m >>= 1) {
    ssx += __shfl_xor(ssx, m, 32);
    ssy += __shfl_xor(ssy, m, 32);
  }

  __shared__ float wx[8], wy[8];
  __shared__ float sx_s, sy_s;
  if (lane == 0) { wx[w] = ssx; wy[w] = ssy; }
  __syncthreads();
  if (tid == 0) {
    float tx = 0.f, ty = 0.f;
#pragma unroll
    for (int i = 0; i < 8; ++i) { tx += wx[i]; ty += wy[i]; }
    sx_s = INV_T / fmaxf(sqrtf(tx), EPSN);   // fold temperature into X side
    sy_s = 1.0f  / fmaxf(sqrtf(ty), EPSN);
    rowsum[row] = 0.0f;                       // deterministic re-init each call
    diag[row]   = 0.0f;
  }
  __syncthreads();
  const float sx = sx_s, sy = sy_s;

  v4h xo, yo;
  xo[0] = (_Float16)(xv.x * sx); xo[1] = (_Float16)(xv.y * sx);
  xo[2] = (_Float16)(xv.z * sx); xo[3] = (_Float16)(xv.w * sx);
  yo[0] = (_Float16)(yv.x * sy); yo[1] = (_Float16)(yv.y * sy);
  yo[2] = (_Float16)(yv.z * sy); yo[3] = (_Float16)(yv.w * sy);
  *(v4h*)(Xn + (size_t)row * DDIM + tid * 4) = xo;
  *(v4h*)(Yn + (size_t)row * DDIM + tid * 4) = yo;
}

// ---------------------------------------------------------------------------
// TDM: issue a tensor_load_to_lds of a (nrows x 32 f16) tile whose rows are
// DDIM-element rows of the source tensor. D# per CDNA5 ISA:
//   group0: [1:0]=count=1, [63:32]=lds_addr, [120:64]=global tile addr,
//           [127:126]=type=2
//   group1: [17:16]=data_size=1 (2B), tensor_dim0=DDIM, tensor_dim1=BDIM,
//           tile_dim0=32, tile_dim1=nrows, tensor_dim0_stride=DDIM
//   groups 2/3 + extra group: zero (2-D tensor)
// This toolchain exposes the 6-arg builtin:
//   (uint32x4 g0, int32x8 g1, int32x4, int32x4, int32x8, i32 cpol)
// ---------------------------------------------------------------------------
__device__ __forceinline__ void tdm_load_tile(const _Float16* gsrc,
                                              unsigned lds_off, int nrows)
{
  const unsigned long long ga = (unsigned long long)(uintptr_t)gsrc;
  v4u g0;
  g0[0] = 1u;                                            // count=1 (valid)
  g0[1] = lds_off;                                       // lds_addr (bytes)
  g0[2] = (unsigned)(ga & 0xFFFFFFFFu);                  // global_addr[31:0]
  g0[3] = (unsigned)((ga >> 32) & 0x01FFFFFFu) | (2u << 30); // addr[56:32]|type=2
  v8i g1;
  g1[0] = (int)(1u << 16);                               // data_size=1 -> 2 bytes
  g1[1] = (int)((DDIM & 0xFFFFu) << 16);                 // tensor_dim0 lo16
  g1[2] = (int)(((DDIM >> 16) & 0xFFFFu) | ((BDIM & 0xFFFFu) << 16)); // dim0 hi | dim1 lo
  g1[3] = (int)(((BDIM >> 16) & 0xFFFFu) | ((unsigned)KSTEP << 16));  // dim1 hi | tile_dim0=32
  g1[4] = (int)(nrows & 0xFFFF);                         // tile_dim1 | tile_dim2=0
  g1[5] = (int)DDIM;                                     // tensor_dim0_stride lo32
  g1[6] = 0;                                             // stride hi | dim1_stride lo
  g1[7] = 0;
  v4i g2 = {0, 0, 0, 0};
  v4i g3 = {0, 0, 0, 0};
  v8i g4 = {0, 0, 0, 0, 0, 0, 0, 0};
  __builtin_amdgcn_tensor_load_to_lds(g0, g1, g2, g3, g4, 0);
}

// ---------------------------------------------------------------------------
// LDS fragment loaders, CDNA5 ISA 7.12.2 layouts (wave32), tile stride = 32 f16.
// ---------------------------------------------------------------------------
__device__ __forceinline__ v16h lds_afrag(const _Float16* base, int tro, int lane) {
  const int m = lane & 15, half = lane >> 4;
  const _Float16* p = base + (tro + m) * KSTEP + half * 8;
  v8h lo = *(const v8h*)p;           // K = 8*half + 0..7
  v8h hi = *(const v8h*)(p + 16);    // K = 16 + 8*half + 0..7
  v16h f;
#pragma unroll
  for (int i = 0; i < 8; ++i) { f[i] = lo[i]; f[i + 8] = hi[i]; }
  return f;
}

__device__ __forceinline__ v16h lds_bfrag(const _Float16* base, int tco, int lane) {
  const int n = lane & 15, half = lane >> 4;
  const _Float16* p = base + (tco + n) * KSTEP + half * 8;
  v8h lo = *(const v8h*)p;           // K = 8*half + 0..7
  v8h hi = *(const v8h*)(p + 16);    // K = 16 + 8*half + 0..7
  v16h f;
#pragma unroll
  for (int i = 0; i < 8; ++i) { f[i] = lo[i]; f[i + 8] = hi[i]; }
  return f;
}

// ---------------------------------------------------------------------------
// Phase 2: fused GEMM + sum-of-exp + diagonal, TDM double-buffered via LDS.
// Block = 256 threads = 8 wave32, 128x128 logit tile; wave = 64x32 (4x2 WMMA).
// ---------------------------------------------------------------------------
__global__ __launch_bounds__(256) void sim_gemm_lse(
    const _Float16* __restrict__ Xn, const _Float16* __restrict__ Yn,
    float* __restrict__ rowsum, float* __restrict__ diag)
{
  __shared__ _Float16 ldsA[2][TILE_M * KSTEP];   // 2 x 8 KB
  __shared__ _Float16 ldsB[2][TILE_N * KSTEP];   // 2 x 8 KB
  __shared__ float    lds_rs[TILE_M];

  const int tid  = threadIdx.x;
  const int w    = tid >> 5;
  const int lane = tid & 31;
  const int wm   = w >> 2;                  // 0..1
  const int wn   = w & 3;                   // 0..3
  const int rowBase = blockIdx.y * TILE_M;
  const int colBase = blockIdx.x * TILE_N;
  const int wRowT = wm * 64;                // tile-relative
  const int wColT = wn * 32;

  const _Float16* gA = Xn + (size_t)rowBase * DDIM;
  const _Float16* gB = Yn + (size_t)colBase * DDIM;

  const unsigned offA[2] = { (unsigned)(uintptr_t)&ldsA[0][0],
                             (unsigned)(uintptr_t)&ldsA[1][0] };
  const unsigned offB[2] = { (unsigned)(uintptr_t)&ldsB[0][0],
                             (unsigned)(uintptr_t)&ldsB[1][0] };

  v8f acc[4][2] = {};

  // prologue: wave 0 kicks off TDM DMA for k-step 0 into buffer 0
  if (w == 0) {
    tdm_load_tile(gA, offA[0], TILE_M);
    tdm_load_tile(gB, offB[0], TILE_N);
  }

  for (int ks = 0; ks < NKSTEP; ++ks) {
    const int buf = ks & 1;
    // prefetch next k-step into the other buffer (safe: it was consumed in the
    // previous iteration and the end-of-iteration barrier has since passed)
    if (w == 0 && ks + 1 < NKSTEP) {
      tdm_load_tile(gA + (size_t)(ks + 1) * KSTEP, offA[buf ^ 1], TILE_M);
      tdm_load_tile(gB + (size_t)(ks + 1) * KSTEP, offB[buf ^ 1], TILE_N);
    }
    // TDM completes in-order per wave: <=2 pending leaves only the prefetch.
    if (ks + 1 < NKSTEP) __builtin_amdgcn_s_wait_tensorcnt(2);
    else                 __builtin_amdgcn_s_wait_tensorcnt(0);
    __syncthreads();                         // buffer `buf` is ready for all

    v16h a[4], b[2];
#pragma unroll
    for (int mi = 0; mi < 4; ++mi) a[mi] = lds_afrag(ldsA[buf], wRowT + mi * 16, lane);
#pragma unroll
    for (int ni = 0; ni < 2; ++ni) b[ni] = lds_bfrag(ldsB[buf], wColT + ni * 16, lane);
#pragma unroll
    for (int mi = 0; mi < 4; ++mi)
#pragma unroll
      for (int ni = 0; ni < 2; ++ni)
        acc[mi][ni] = __builtin_amdgcn_wmma_f32_16x16x32_f16(
            false, a[mi], false, b[ni], (short)0, acc[mi][ni], false, false);

    __syncthreads();                         // done reading `buf`; may overwrite
  }

  // ---- epilogue: exp + per-row reduction ----
  for (int i = tid; i < TILE_M; i += 256) lds_rs[i] = 0.0f;
  __syncthreads();

  const int nlo  = lane & 15;
  const int half = lane >> 4;
#pragma unroll
  for (int mi = 0; mi < 4; ++mi) {
#pragma unroll
    for (int r = 0; r < 8; ++r) {
      const int grow = rowBase + wRowT + mi * 16 + r + 8 * half;  // M = r + 8*half
      float esum = 0.0f;
#pragma unroll
      for (int ni = 0; ni < 2; ++ni) {
        const float v = acc[mi][ni][r];
        const int gcol = colBase + wColT + ni * 16 + nlo;         // N = lane%16
        if (gcol == grow) diag[grow] = v;                         // unique writer
        esum += __expf(v);
      }
#pragma unroll
      for (int m = 8; m >= 1; m >>= 1) esum += __shfl_xor(esum, m, 32);
      if (nlo == 0) atomicAdd(&lds_rs[grow - rowBase], esum);     // ds_add_f32
    }
  }
  __syncthreads();
  for (int i = tid; i < TILE_M; i += 256)
    atomicAdd(&rowsum[rowBase + i], lds_rs[i]);   // one global atomic/row/block
}

// ---------------------------------------------------------------------------
// Phase 3: loss = mean(log(rowsum) - diag)
// ---------------------------------------------------------------------------
__global__ __launch_bounds__(256) void finalize_loss(
    const float* __restrict__ rowsum, const float* __restrict__ diag,
    float* __restrict__ out)
{
  const int tid  = threadIdx.x;
  const int w    = tid >> 5;
  const int lane = tid & 31;
  float s = 0.0f;
  for (int i = tid; i < BDIM; i += 256)
    s += __logf(rowsum[i]) - diag[i];
#pragma unroll
  for (int m = 16; m >= 1; m >>= 1) s += __shfl_xor(s, m, 32);
  __shared__ float ws_[8];
  if (lane == 0) ws_[w] = s;
  __syncthreads();
  if (tid == 0) {
    float t = 0.0f;
#pragma unroll
    for (int i = 0; i < 8; ++i) t += ws_[i];
    out[0] = t / (float)BDIM;
  }
}

// ---------------------------------------------------------------------------
extern "C" void kernel_launch(void* const* d_in, const int* in_sizes, int n_in,
                              void* d_out, int out_size, void* d_ws, size_t ws_size,
                              hipStream_t stream) {
  (void)in_sizes; (void)n_in; (void)out_size; (void)ws_size;
  const float* X = (const float*)d_in[0];
  const float* Y = (const float*)d_in[1];

  _Float16* Xn = (_Float16*)d_ws;                      // 16 MB
  _Float16* Yn = Xn + (size_t)BDIM * DDIM;             // 16 MB
  float* rowsum = (float*)(Yn + (size_t)BDIM * DDIM);  // 32 KB
  float* diag   = rowsum + BDIM;                       // 32 KB

  normalize_rows<<<BDIM, 256, 0, stream>>>(X, Y, Xn, Yn, rowsum, diag);
  dim3 grid(BDIM / TILE_N, BDIM / TILE_M);
  sim_gemm_lse<<<grid, 256, 0, stream>>>(Xn, Yn, rowsum, diag);
  finalize_loss<<<1, 256, 0, stream>>>(rowsum, diag, (float*)d_out);
}
